// CrossAttention_10574209482885
// MI455X (gfx1250) — compile-verified, matrix operand order checked by reading, not run on hardware
//
#include <hip/hip_runtime.h>
#include <hip/hip_bf16.h>
#include <math.h>

typedef __bf16 bf16;
typedef __attribute__((ext_vector_type(16))) __bf16 v16bf;
typedef __attribute__((ext_vector_type(8)))  __bf16 v8bf;
typedef __attribute__((ext_vector_type(8)))  float   v8f;

#define LN_EPS  1e-5f
#define NRM_EPS 1e-12f

#define BM 128
#define BN 64
#define BK 32

// ---------------------------------------------------------------------------
// LayerNorm over last dim (C). One block per row. Writes fp32 and/or bf16.
// ---------------------------------------------------------------------------
__global__ __launch_bounds__(256)
void ln_kernel(const float* __restrict__ x, const float* __restrict__ w,
               const float* __restrict__ bvec, float* __restrict__ outf,
               bf16* __restrict__ outb, int C) {
  __shared__ float s1[256], s2[256];
  const int tid = threadIdx.x;
  const size_t base = (size_t)blockIdx.x * C;
  float s = 0.f, sq = 0.f;
  for (int c = tid; c < C; c += 256) {
    float v = x[base + c];
    s += v; sq += v * v;
  }
  s1[tid] = s; s2[tid] = sq;
  __syncthreads();
  for (int o = 128; o > 0; o >>= 1) {
    if (tid < o) { s1[tid] += s1[tid + o]; s2[tid] += s2[tid + o]; }
    __syncthreads();
  }
  const float mu  = s1[0] / (float)C;
  const float var = s2[0] / (float)C - mu * mu;
  const float inv = rsqrtf(var + LN_EPS);
  for (int c = tid; c < C; c += 256) {
    float v = (x[base + c] - mu) * inv * w[c] + bvec[c];
    if (outf) outf[base + c] = v;
    if (outb) outb[base + c] = (bf16)v;
  }
}

// ---------------------------------------------------------------------------
// fp32 -> bf16 (weights)
// ---------------------------------------------------------------------------
__global__ void to_bf16_kernel(const float* __restrict__ src, bf16* __restrict__ dst, size_t n) {
  for (size_t i = (size_t)blockIdx.x * blockDim.x + threadIdx.x; i < n;
       i += (size_t)gridDim.x * blockDim.x)
    dst[i] = (bf16)src[i];
}

// ---------------------------------------------------------------------------
// Row squared-norm of bf16 matrix (ld = C). One block per row.
// ---------------------------------------------------------------------------
__global__ __launch_bounds__(256)
void rowsq_kernel(const bf16* __restrict__ x, float* __restrict__ out, int C) {
  __shared__ float s1[256];
  const int tid = threadIdx.x;
  const size_t base = (size_t)blockIdx.x * C;
  float s = 0.f;
  for (int c = tid; c < C; c += 256) {
    float v = (float)x[base + c];
    s += v * v;
  }
  s1[tid] = s;
  __syncthreads();
  for (int o = 128; o > 0; o >>= 1) {
    if (tid < o) s1[tid] += s1[tid + o];
    __syncthreads();
  }
  if (tid == 0) out[blockIdx.x] = s1[0];
}

// ---------------------------------------------------------------------------
// Fragment load from a k-major LDS tile T[row][BK] for v_wmma_*_16x16x32:
// lane half h holds K {8h..8h+7} in regs 0-3 and K {16+8h..16+8h+7} in 4-7,
// i.e. two contiguous 16-byte runs -> two ds_load_b128.
// ---------------------------------------------------------------------------
__device__ __forceinline__ v16bf load_frag(const bf16* row, int half) {
  const v8bf lo = *(const v8bf*)(row + half * 8);
  const v8bf hi = *(const v8bf*)(row + 16 + half * 8);
  v16bf f;
#pragma unroll
  for (int e = 0; e < 8; ++e) { f[e] = lo[e]; f[8 + e] = hi[e]; }
  return f;
}

// ---------------------------------------------------------------------------
// C = A(MxK,bf16) * B(KxN,bf16) + bias, optional exact-GELU, optional residual.
// split>0: cols<split -> bf16 out (Cb), cols>=split -> f32 out (Cf, col-split).
// Tiles: 128x64x32; 8 waves in 4x2; each wave a 32x32 sub-tile (4 WMMAs/step).
// A and B both stored k-major in LDS so every fragment is 2x ds_load_b128.
// ---------------------------------------------------------------------------
__global__ __launch_bounds__(256)
void gemm_bf16_nn(const bf16* __restrict__ A, int lda,
                  const bf16* __restrict__ Bm, int ldb,
                  const float* __restrict__ bias,
                  float* __restrict__ Cf, int ldc,
                  bf16* __restrict__ Cb, int ldcb,
                  const float* __restrict__ res, int ldr,
                  int M, int Nn, int K, int act, int split) {
  __shared__ bf16 As[BM][BK];   // k-major
  __shared__ bf16 Bs[BN][BK];   // k-major (transposed from global)
  const int tid = threadIdx.x;
  const int wid = tid >> 5, lane = tid & 31;
  const int half = lane >> 4, l16 = lane & 15;
  const int m0 = blockIdx.y * BM, n0 = blockIdx.x * BN;
  const int wm = (wid & 3) * 32;
  const int wn = (wid >> 2) * 32;
  v8f acc[2][2];
#pragma unroll
  for (int mi = 0; mi < 2; ++mi)
#pragma unroll
    for (int ni = 0; ni < 2; ++ni)
      acc[mi][ni] = (v8f){0.f,0.f,0.f,0.f,0.f,0.f,0.f,0.f};

  for (int k0 = 0; k0 < K; k0 += BK) {
    // A tile: 128x32 = 4096 bf16, two b128 passes
#pragma unroll
    for (int t = 0; t < 2; ++t) {
      int lin = (tid + t * 256) * 8;
      int r = lin >> 5, c = lin & 31;
      *(v8bf*)&As[r][c] = *(const v8bf*)&A[(size_t)(m0 + r) * lda + (k0 + c)];
    }
    // B tile: 64x32, read n-contiguous b128, scatter to k-major LDS
    {
      int lin = tid * 8;
      int n = lin & 63, k = lin >> 6;
      v8bf bv = *(const v8bf*)&Bm[(size_t)(k0 + k) * ldb + (n0 + n)];
#pragma unroll
      for (int i = 0; i < 8; ++i) Bs[n + i][k] = bv[i];
    }
    if (k0 + BK < K) {
      __builtin_prefetch(&A[(size_t)(m0 + (tid >> 1)) * lda + (k0 + BK)], 0, 1);
      __builtin_prefetch(&Bm[(size_t)(k0 + BK + (tid >> 3)) * ldb + n0], 0, 1);
    }
    __syncthreads();
    v16bf af0 = load_frag(&As[wm +      l16][0], half);
    v16bf af1 = load_frag(&As[wm + 16 + l16][0], half);
    v16bf bf0 = load_frag(&Bs[wn +      l16][0], half);
    v16bf bf1 = load_frag(&Bs[wn + 16 + l16][0], half);
    acc[0][0] = __builtin_amdgcn_wmma_f32_16x16x32_bf16(false, af0, false, bf0, (short)0, acc[0][0], false, false);
    acc[0][1] = __builtin_amdgcn_wmma_f32_16x16x32_bf16(false, af0, false, bf1, (short)0, acc[0][1], false, false);
    acc[1][0] = __builtin_amdgcn_wmma_f32_16x16x32_bf16(false, af1, false, bf0, (short)0, acc[1][0], false, false);
    acc[1][1] = __builtin_amdgcn_wmma_f32_16x16x32_bf16(false, af1, false, bf1, (short)0, acc[1][1], false, false);
    __syncthreads();
  }

#pragma unroll
  for (int mi = 0; mi < 2; ++mi)
#pragma unroll
    for (int ni = 0; ni < 2; ++ni)
#pragma unroll
      for (int r = 0; r < 8; ++r) {
        const int row = m0 + wm + mi * 16 + r + 8 * half;
        const int col = n0 + wn + ni * 16 + l16;
        float val = acc[mi][ni][r];
        if (bias) val += bias[col];
        if (act) val = 0.5f * val * (1.f + erff(val * 0.70710678118654752f));
        if (split > 0) {
          if (col < split) { if (Cb) Cb[(size_t)row * ldcb + col] = (bf16)val; }
          else             { if (Cf) Cf[(size_t)row * ldc + (col - split)] = val; }
        } else {
          if (res) val += res[(size_t)row * ldr + col];
          if (Cf) Cf[(size_t)row * ldc + col] = val;
          if (Cb) Cb[(size_t)row * ldcb + col] = (bf16)val;
        }
      }
}

// ---------------------------------------------------------------------------
// Pairwise Euclidean distance: S[b,i,j] = sqrt(max(|Xi|^2+|Yj|^2-2Xi.Yj, eps))
// X,Y bf16 row-major (slab base, rows b*N+i, ld=K). One batch per blockIdx.z.
// Both LDS tiles are k-major; Y tile copy is fully contiguous b128 in/out.
// ---------------------------------------------------------------------------
__global__ __launch_bounds__(256)
void cdist_bf16_nt(const bf16* __restrict__ X, const bf16* __restrict__ Y,
                   const float* __restrict__ xsq, const float* __restrict__ ysq,
                   float* __restrict__ S, int Nn, int K) {
  __shared__ bf16 As[BM][BK];
  __shared__ bf16 Bs[BN][BK];
  const int tid = threadIdx.x;
  const int wid = tid >> 5, lane = tid & 31;
  const int half = lane >> 4, l16 = lane & 15;
  const int b = blockIdx.z;
  const int m0 = blockIdx.y * BM, n0 = blockIdx.x * BN;
  const size_t xr0 = (size_t)b * Nn + m0;
  const size_t yr0 = (size_t)b * Nn + n0;
  const int wm = (wid & 3) * 32;
  const int wn = (wid >> 2) * 32;
  v8f acc[2][2];
#pragma unroll
  for (int mi = 0; mi < 2; ++mi)
#pragma unroll
    for (int ni = 0; ni < 2; ++ni)
      acc[mi][ni] = (v8f){0.f,0.f,0.f,0.f,0.f,0.f,0.f,0.f};

  for (int k0 = 0; k0 < K; k0 += BK) {
#pragma unroll
    for (int t = 0; t < 2; ++t) {
      int lin = (tid + t * 256) * 8;
      int r = lin >> 5, c = lin & 31;
      *(v8bf*)&As[r][c] = *(const v8bf*)&X[(xr0 + r) * K + (k0 + c)];
    }
    {
      int lin = tid * 8;
      int jj = lin >> 5, kk = lin & 31;
      *(v8bf*)&Bs[jj][kk] = *(const v8bf*)&Y[(yr0 + jj) * K + (k0 + kk)];
    }
    if (k0 + BK < K) {
      __builtin_prefetch(&X[(xr0 + (tid >> 1)) * K + (k0 + BK)], 0, 1);
      __builtin_prefetch(&Y[(yr0 + (tid >> 2)) * K + (k0 + BK)], 0, 1);
    }
    __syncthreads();
    v16bf af0 = load_frag(&As[wm +      l16][0], half);
    v16bf af1 = load_frag(&As[wm + 16 + l16][0], half);
    v16bf bf0 = load_frag(&Bs[wn +      l16][0], half);
    v16bf bf1 = load_frag(&Bs[wn + 16 + l16][0], half);
    acc[0][0] = __builtin_amdgcn_wmma_f32_16x16x32_bf16(false, af0, false, bf0, (short)0, acc[0][0], false, false);
    acc[0][1] = __builtin_amdgcn_wmma_f32_16x16x32_bf16(false, af0, false, bf1, (short)0, acc[0][1], false, false);
    acc[1][0] = __builtin_amdgcn_wmma_f32_16x16x32_bf16(false, af1, false, bf0, (short)0, acc[1][0], false, false);
    acc[1][1] = __builtin_amdgcn_wmma_f32_16x16x32_bf16(false, af1, false, bf1, (short)0, acc[1][1], false, false);
    __syncthreads();
  }

#pragma unroll
  for (int mi = 0; mi < 2; ++mi)
#pragma unroll
    for (int r = 0; r < 8; ++r) {
      const int row = m0 + wm + mi * 16 + r + 8 * half;
      const float xs = xsq[(size_t)b * Nn + row];
#pragma unroll
      for (int ni = 0; ni < 2; ++ni) {
        const int col = n0 + wn + ni * 16 + l16;
        float d2 = xs + ysq[(size_t)b * Nn + col] - 2.f * acc[mi][ni][r];
        S[((size_t)b * Nn + row) * Nn + col] = sqrtf(fmaxf(d2, NRM_EPS));
      }
    }
}

// ---------------------------------------------------------------------------
// Global min/max reduce (all values >= 0 -> uint bit order == float order).
// ---------------------------------------------------------------------------
__global__ void init_scalars_kernel(unsigned int* g) {
  g[0] = 0x7F800000u;  // +inf (min slot)
  g[1] = 0u;           // 0    (max slot)
}

__global__ __launch_bounds__(256)
void reduce_minmax_kernel(const float* __restrict__ S, size_t n, unsigned int* __restrict__ g) {
  __shared__ float smn[256], smx[256];
  const int tid = threadIdx.x;
  float lo = 3.0e38f, hi = 0.f;
  for (size_t i = (size_t)blockIdx.x * 256 + tid; i < n; i += (size_t)gridDim.x * 256) {
    float v = S[i];
    lo = fminf(lo, v); hi = fmaxf(hi, v);
  }
  smn[tid] = lo; smx[tid] = hi;
  __syncthreads();
  for (int o = 128; o > 0; o >>= 1) {
    if (tid < o) { smn[tid] = fminf(smn[tid], smn[tid + o]); smx[tid] = fmaxf(smx[tid], smx[tid + o]); }
    __syncthreads();
  }
  if (tid == 0) {
    atomicMin(g + 0, __float_as_uint(smn[0]));
    atomicMax(g + 1, __float_as_uint(smx[0]));
  }
}

// ---------------------------------------------------------------------------
// Row-direction reduce over j: f = mode ? (gmax - v) : (v - gmin).
// do_sq: out[b,i] = sqrt(sum_j f^2); else out[b,i] = sum_j f / max(div[b,j],eps)
// ---------------------------------------------------------------------------
__global__ __launch_bounds__(256)
void row_reduce_kernel(const float* __restrict__ S, const unsigned int* __restrict__ gb,
                       int mode, const float* __restrict__ divv,
                       float* __restrict__ out, int Nn, int do_sq) {
  __shared__ float s1[256];
  const int tid = threadIdx.x;
  const int rowg = blockIdx.x;            // b*N + i
  const int b = rowg / Nn;
  const size_t base = (size_t)rowg * Nn;
  const float g = __uint_as_float(gb[mode]);
  float s = 0.f;
  for (int j = tid; j < Nn; j += 256) {
    float v = S[base + j];
    float f = mode ? (g - v) : (v - g);
    if (do_sq) s += f * f;
    else       s += f / fmaxf(divv[(size_t)b * Nn + j], NRM_EPS);
  }
  s1[tid] = s;
  __syncthreads();
  for (int o = 128; o > 0; o >>= 1) {
    if (tid < o) s1[tid] += s1[tid + o];
    __syncthreads();
  }
  if (tid == 0) out[rowg] = do_sq ? sqrtf(s1[0]) : s1[0];
}

// ---------------------------------------------------------------------------
// Column-direction reduce over i (coalesced across j threads).
// ---------------------------------------------------------------------------
__global__ __launch_bounds__(256)
void col_reduce_kernel(const float* __restrict__ S, const unsigned int* __restrict__ gb,
                       int mode, const float* __restrict__ divv,
                       float* __restrict__ out, int Nn, int do_sq) {
  const int j = blockIdx.x * 256 + threadIdx.x;
  const int b = blockIdx.y;
  const float g = __uint_as_float(gb[mode]);
  float s = 0.f;
  for (int i = 0; i < Nn; ++i) {
    float v = S[((size_t)b * Nn + i) * Nn + j];
    float f = mode ? (g - v) : (v - g);
    if (do_sq) s += f * f;
    else       s += f / fmaxf(divv[(size_t)b * Nn + i], NRM_EPS);
  }
  out[(size_t)b * Nn + j] = do_sq ? sqrtf(s) : s;
}

// ---------------------------------------------------------------------------
// map = base + (tanh(theta)+1) * (sig(al)*(d11+d12)/2 + (1-sig(al))*(d21+d22)/2)
// ---------------------------------------------------------------------------
__global__ void combine_maps_kernel(const float* __restrict__ base,
                                    const float* __restrict__ d11, const float* __restrict__ d12,
                                    const float* __restrict__ d21, const float* __restrict__ d22,
                                    const float* __restrict__ theta, const float* __restrict__ alpha,
                                    float* __restrict__ out, int n) {
  int i = blockIdx.x * 256 + threadIdx.x;
  if (i >= n) return;
  const float k2 = tanhf(theta[0]) + 1.f;
  const float al = 1.f / (1.f + expf(-alpha[0]));
  out[i] = base[i] + k2 * (al * 0.5f * (d11[i] + d12[i]) + (1.f - al) * 0.5f * (d21[i] + d22[i]));
}

// ---------------------------------------------------------------------------
// mode 0: o = map*v + ln  (anchor/positive); mode 1: o = map*v*ln  (negatives)
// In-place on ln slab.
// ---------------------------------------------------------------------------
__global__ __launch_bounds__(256)
void apply_attn_kernel(const float* __restrict__ mapv, const float* __restrict__ v,
                       float* __restrict__ ln_io, int C, int mode) {
  const size_t idx = (size_t)blockIdx.x * 256 + threadIdx.x;
  const float m = mapv[idx / C];
  const float l = ln_io[idx];
  ln_io[idx] = mode ? (m * v[idx] * l) : (m * v[idx] + l);
}

// ===========================================================================
extern "C" void kernel_launch(void* const* d_in, const int* in_sizes, int n_in,
                              void* d_out, int out_size, void* d_ws, size_t ws_size,
                              hipStream_t stream) {
  (void)in_sizes; (void)n_in; (void)out_size; (void)ws_size;
  const int Bb = 8, Nn = 1024, Cc = 768;
  const int slab = Bb * Nn;                 // rows per slab = 8192
  const size_t slabE = (size_t)slab * Cc;   // elements per slab
  const int R = 6 * slab;                   // 49152 rows total

  const float* anchor   = (const float*)d_in[0];
  const float* positive = (const float*)d_in[1];
  const float* neg1     = (const float*)d_in[2];
  const float* neg2     = (const float*)d_in[3];
  const float* n1w = (const float*)d_in[4];
  const float* n1b = (const float*)d_in[5];
  const float* n2w = (const float*)d_in[6];
  const float* n2b = (const float*)d_in[7];
  const float* projW = (const float*)d_in[8];
  const float* projB = (const float*)d_in[9];
  const float* fc1W  = (const float*)d_in[10];
  const float* fc1B  = (const float*)d_in[11];
  const float* fc2W  = (const float*)d_in[12];
  const float* fc2B  = (const float*)d_in[13];
  const float* theta = (const float*)d_in[14];
  const float* alpha = (const float*)d_in[15];
  float* dout = (float*)d_out;

  // ---- bump allocator over d_ws ----
  char* ws = (char*)d_ws;
  size_t off = 0;
  auto alloc = [&](size_t bytes) -> void* {
    off = (off + 255) & ~(size_t)255;
    void* p = ws + off;
    off += bytes;
    return p;
  };
  float* lnf   = (float*)alloc(6 * slabE * sizeof(float));   // LN1 out / o tensors
  bf16*  lnb   = (bf16*) alloc(6 * slabE * sizeof(bf16));    // LN1/LN2 bf16
  bf16*  qkb   = (bf16*) alloc(6 * slabE * sizeof(bf16));    // qk (proj cols < 768)
  float* vf    = (float*)alloc(6 * slabE * sizeof(float));   // v  (proj cols >= 768)
  bf16*  h1b   = (bf16*) alloc((size_t)slab * 4 * Cc * sizeof(bf16)); // MLP hidden (per-slab reuse)
  bf16*  pWb   = (bf16*) alloc((size_t)Cc * 2 * Cc * sizeof(bf16));
  bf16*  f1Wb  = (bf16*) alloc((size_t)Cc * 4 * Cc * sizeof(bf16));
  bf16*  f2Wb  = (bf16*) alloc((size_t)4 * Cc * Cc * sizeof(bf16));
  float* S     = (float*)alloc((size_t)Bb * Nn * Nn * sizeof(float)); // one cdist matrix (reused)
  float* xsq   = (float*)alloc((size_t)R * sizeof(float));
  float* rn    = (float*)alloc((size_t)Bb * Nn * sizeof(float));
  float* cn    = (float*)alloc((size_t)Bb * Nn * sizeof(float));
  float* Sa    = (float*)alloc((size_t)Bb * Nn * sizeof(float));
  float* Sp1   = (float*)alloc((size_t)Bb * Nn * sizeof(float));
  float* dd0   = (float*)alloc((size_t)Bb * Nn * sizeof(float));
  float* dd1   = (float*)alloc((size_t)Bb * Nn * sizeof(float));
  float* dd2   = (float*)alloc((size_t)Bb * Nn * sizeof(float));
  float* dd3   = (float*)alloc((size_t)Bb * Nn * sizeof(float));
  float* maps[6];
  for (int s = 0; s < 6; ++s) maps[s] = (float*)alloc((size_t)Bb * Nn * sizeof(float));
  unsigned int* gb = (unsigned int*)alloc(2 * sizeof(unsigned int));

  // ---- weights to bf16 ----
  to_bf16_kernel<<<1024, 256, 0, stream>>>(projW, pWb, (size_t)Cc * 2 * Cc);
  to_bf16_kernel<<<1024, 256, 0, stream>>>(fc1W,  f1Wb, (size_t)Cc * 4 * Cc);
  to_bf16_kernel<<<1024, 256, 0, stream>>>(fc2W,  f2Wb, (size_t)4 * Cc * Cc);

  // ---- LayerNorm 1 on the six slabs ----
  const float* src[6] = { anchor, positive, neg1, neg1 + slabE, neg2, neg2 + slabE };
  for (int s = 0; s < 6; ++s)
    ln_kernel<<<slab, 256, 0, stream>>>(src[s], n1w, n1b, lnf + s * slabE, lnb + s * slabE, Cc);

  // ---- projection GEMM: (R x 768) @ (768 x 1536); split into qk(bf16)/v(f32)
  gemm_bf16_nn<<<dim3(2 * Cc / BN, R / BM), 256, 0, stream>>>(
      lnb, Cc, pWb, 2 * Cc, projB, vf, Cc, qkb, Cc, nullptr, 0,
      R, 2 * Cc, Cc, /*act=*/0, /*split=*/Cc);

  // ---- qk row squared norms (all 6 slabs) ----
  rowsq_kernel<<<R, 256, 0, stream>>>(qkb, xsq, Cc);

  // ---- attention maps ----
  const size_t nS = (size_t)Bb * Nn * Nn;
  auto run_cdist = [&](int sx, int sy) {
    cdist_bf16_nt<<<dim3(Nn / BN, Nn / BM, Bb), 256, 0, stream>>>(
        qkb + (size_t)sx * slabE, qkb + (size_t)sy * slabE,
        xsq + (size_t)sx * slab, xsq + (size_t)sy * slab, S, Nn, Cc);
  };
  auto run_reduce = [&]() {
    init_scalars_kernel<<<1, 1, 0, stream>>>(gb);
    reduce_minmax_kernel<<<2048, 256, 0, stream>>>(S, nS, gb);
  };
  auto proc_d = [&](int sx, int sy, float* dst) {
    run_cdist(sx, sy);
    run_reduce();
    row_reduce_kernel<<<Bb * Nn, 256, 0, stream>>>(S, gb, 0, nullptr, rn, Nn, 1);
    col_reduce_kernel<<<dim3(Nn / 256, Bb), 256, 0, stream>>>(S, gb, 0, rn, dst, Nn, 0);
  };
  auto attn_call = [&](int sa_, int sp_, int s1a, int s1b, int s2a, int s2b,
                       float* m1, float* m2) {
    // S_p = gmax - cdist(positive, anchor)
    run_cdist(sp_, sa_);
    run_reduce();
    row_reduce_kernel<<<Bb * Nn, 256, 0, stream>>>(S, gb, 1, nullptr, rn, Nn, 1);
    col_reduce_kernel<<<dim3(Nn / 256, Bb), 256, 0, stream>>>(S, gb, 1, nullptr, cn, Nn, 1);
    col_reduce_kernel<<<dim3(Nn / 256, Bb), 256, 0, stream>>>(S, gb, 1, rn, Sa, Nn, 0);
    row_reduce_kernel<<<Bb * Nn, 256, 0, stream>>>(S, gb, 1, cn, Sp1, Nn, 0);
    // negatives vs anchor -> map1
    proc_d(s1a, sa_, dd0); proc_d(s1b, sa_, dd1);
    proc_d(s2a, sa_, dd2); proc_d(s2b, sa_, dd3);
    combine_maps_kernel<<<(Bb * Nn + 255) / 256, 256, 0, stream>>>(
        Sa, dd0, dd1, dd2, dd3, theta, alpha, m1, Bb * Nn);
    // negatives vs positive -> map2
    proc_d(s1a, sp_, dd0); proc_d(s1b, sp_, dd1);
    proc_d(s2a, sp_, dd2); proc_d(s2b, sp_, dd3);
    combine_maps_kernel<<<(Bb * Nn + 255) / 256, 256, 0, stream>>>(
        Sp1, dd0, dd1, dd2, dd3, theta, alpha, m2, Bb * Nn);
  };
  // slabs: 0=anchor 1=positive 2=neg1[0] 3=neg1[1] 4=neg2[0] 5=neg2[1]
  attn_call(0, 1, 2, 3, 4, 5, maps[0], maps[1]);  // a_a, a_p
  attn_call(2, 3, 0, 1, 4, 5, maps[2], maps[3]);  // n1_a, n1_p
  attn_call(4, 5, 0, 1, 2, 3, maps[4], maps[5]);  // n2_a, n2_p

  // ---- apply attention (in-place on lnf -> "o" tensors) ----
  const int modes[6] = {0, 0, 1, 1, 1, 1};
  for (int s = 0; s < 6; ++s)
    apply_attn_kernel<<<(unsigned)(slabE / 256), 256, 0, stream>>>(
        maps[s], vf + s * slabE, lnf + s * slabE, Cc, modes[s]);

  // ---- MLP block per slab: LN2 -> fc1+GELU -> fc2 + bias + residual -> out
  for (int s = 0; s < 6; ++s) {
    ln_kernel<<<slab, 256, 0, stream>>>(lnf + s * slabE, n2w, n2b,
                                        nullptr, lnb + s * slabE, Cc);
    gemm_bf16_nn<<<dim3(4 * Cc / BN, slab / BM), 256, 0, stream>>>(
        lnb + s * slabE, Cc, f1Wb, 4 * Cc, fc1B,
        nullptr, 0, h1b, 4 * Cc, nullptr, 0,
        slab, 4 * Cc, Cc, /*act=gelu*/1, 0);
    gemm_bf16_nn<<<dim3(Cc / BN, slab / BM), 256, 0, stream>>>(
        h1b, 4 * Cc, f2Wb, Cc, fc2B,
        dout + s * slabE, Cc, nullptr, 0, lnf + s * slabE, Cc,
        slab, Cc, 4 * Cc, 0, 0);
  }
}
